// GCNConv_s_944892805297
// MI455X (gfx1250) — matrix operand order repaired and checked by reading, not compiled, not run on hardware
//
#include <hip/hip_runtime.h>
#include <hip/hip_bf16.h>
#include <stdint.h>

typedef __attribute__((ext_vector_type(16))) __bf16   v16bf;
typedef __attribute__((ext_vector_type(8)))  float    v8f;
typedef __attribute__((ext_vector_type(4)))  float    v4f;
typedef __attribute__((ext_vector_type(4)))  unsigned v4u;

#define A_STRIDE 40   // bf16 elems per LDS row: 32 + 8 pad = 80B (16B aligned, conflict-free)
#define B_STRIDE 40

union FragAB {
    v16bf v;
    v4u   q[2];
    unsigned u[8];
    unsigned short h[16];
};

static __device__ __forceinline__ unsigned short f32_to_bf16_rne(float f) {
    unsigned u = __float_as_uint(f);
    u += 0x7FFFu + ((u >> 16) & 1u);          // round to nearest even
    return (unsigned short)(u >> 16);
}

// packed f32x2 -> bf16x2 in ONE instruction (CDNA packed convert)
static __device__ __forceinline__ unsigned pack_bf16x2(float lo, float hi) {
    unsigned u;
    asm("v_cvt_pk_bf16_f32 %0, %1, %2" : "=v"(u) : "v"(lo), "v"(hi));
    return u;
}

// ---------------------------------------------------------------------------
// Pass 1: d[i] = rsqrt(rowsum(A) + l), 0 if degree <= 0
// ---------------------------------------------------------------------------
__global__ __launch_bounds__(256)
void deg_rsqrt_kernel(const float* __restrict__ A, const int* __restrict__ lptr,
                      float* __restrict__ dvec, int N) {
    const int row = blockIdx.x;
    const v4f* a = (const v4f*)(A + (size_t)row * N);
    const int n4 = N >> 2;
    float s = 0.f;
    for (int j = threadIdx.x; j < n4; j += blockDim.x) {
        v4f v = __builtin_nontemporal_load(a + j);    // NT: zero-reuse stream
        s += (v[0] + v[1]) + (v[2] + v[3]);
    }
    for (int off = 16; off > 0; off >>= 1) s += __shfl_down(s, off, 32);
    __shared__ float red[8];
    const int lane = threadIdx.x & 31, w = threadIdx.x >> 5;
    if (lane == 0) red[w] = s;
    __syncthreads();
    if (threadIdx.x == 0) {
        float t = 0.f;
        for (int i = 0; i < 8; i++) t += red[i];
        t += (float)(*lptr);
        dvec[row] = (t > 0.f) ? rsqrtf(t) : 0.f;
    }
}

// ---------------------------------------------------------------------------
// LDS fragment loaders (pass 2 only), wave32 bf16 16x16x32, contiguous k = 16*(lane/16) + s in both A and B
// ---------------------------------------------------------------------------
static __device__ __forceinline__ void load_a_frag(FragAB& f, const unsigned short* Alds,
                                                   int mbase, int lane) {
    const int m = lane & 15, g = lane >> 4;
    const unsigned short* rp = Alds + (mbase + m) * A_STRIDE;
    f.q[0] = *(const v4u*)(rp + 16 * g);
    f.q[1] = *(const v4u*)(rp + 16 * g + 8);
}

static __device__ __forceinline__ void load_b_frag(FragAB& f, const unsigned short* Blds,
                                                   int nbase, int lane) {
    const int n = lane & 15, g = lane >> 4;
    const unsigned short* cp = Blds + (nbase + n) * B_STRIDE;
    f.q[0] = *(const v4u*)(cp + 16 * g);
    f.q[1] = *(const v4u*)(cp + 16 * g + 8);
}

// ---------------------------------------------------------------------------
// Pass 2: supT[col][j] = d[j] * (X @ W)[j][col], stored bf16 COLUMN-MAJOR.
// 4 MB result -> L2-resident; column-major makes pass-3 B fragments two
// contiguous global_load_b128 per lane. Block: 64 rows x 256 cols, 8 waves.
// ---------------------------------------------------------------------------
__global__ __launch_bounds__(256)
void support_gemm_kernel(const float* __restrict__ X, const float* __restrict__ W,
                         const float* __restrict__ dvec, unsigned short* __restrict__ supT,
                         int N, int DIN, int DOUT) {
    __shared__ unsigned short Alds[64 * A_STRIDE];
    __shared__ unsigned short Blds[256 * B_STRIDE];   // col-major [col][k]

    const int tid  = threadIdx.x;
    const int lane = tid & 31;
    const int wave = tid >> 5;
    const int wm   = wave & 3;       // M sub-tile (16 rows each)
    const int wn   = wave >> 2;      // N half (128 cols each)
    const int row0 = blockIdx.x * 64;

    v8f acc[8] = {};

    const int c2 = (tid & 15) * 2;   // A-tile column pair
    const int r0 = tid >> 4;         // A-tile row group (0..15)

    for (int kk = 0; kk < DIN; kk += 32) {
        // stage A tile: X[row0..row0+63][kk..kk+31] -> bf16 LDS
        for (int i = 0; i < 4; i++) {
            int r = r0 + 16 * i;
            const float2 x = *(const float2*)(X + (size_t)(row0 + r) * DIN + kk + c2);
            *(unsigned*)&Alds[r * A_STRIDE + c2] = pack_bf16x2(x.x, x.y);
        }
        // stage B tile: W[kk..kk+31][0..255] transposed to col-major bf16
        {
            const int k  = tid >> 3;
            const int c0 = (tid & 7) * 32;
            const float* wrow = W + (size_t)(kk + k) * DOUT;
            for (int i = 0; i < 8; i++) {
                int c = c0 + 4 * i;
                float4 wv = *(const float4*)(wrow + c);
                Blds[(c + 0) * B_STRIDE + k] = f32_to_bf16_rne(wv.x);
                Blds[(c + 1) * B_STRIDE + k] = f32_to_bf16_rne(wv.y);
                Blds[(c + 2) * B_STRIDE + k] = f32_to_bf16_rne(wv.z);
                Blds[(c + 3) * B_STRIDE + k] = f32_to_bf16_rne(wv.w);
            }
        }
        __syncthreads();

        // preload ALL fragments, then issue 8 back-to-back WMMAs (one dscnt wait)
        FragAB af, bf[8];
        load_a_frag(af, Alds, wm * 16, lane);
        for (int n = 0; n < 8; n++) load_b_frag(bf[n], Blds, wn * 128 + n * 16, lane);
        for (int n = 0; n < 8; n++)
            acc[n] = __builtin_amdgcn_wmma_f32_16x16x32_bf16(
                false, af.v, false, bf[n].v, (short)0, acc[n], false, false);
        __syncthreads();
    }

    // epilogue: scale by d[row], write bf16 column-major; consecutive rows are
    // contiguous in supT -> pack row pairs into one b32 store
    const int g = lane >> 4, nl = lane & 15;
    for (int n = 0; n < 8; n++) {
        int col = wn * 128 + n * 16 + nl;
        unsigned short* cbase = supT + (size_t)col * N + row0 + wm * 16 + g * 8;
        for (int r = 0; r < 8; r += 2) {
            int row = row0 + wm * 16 + g * 8 + r;
            unsigned pk = pack_bf16x2(dvec[row] * acc[n][r],
                                      dvec[row + 1] * acc[n][r + 1]);
            *(unsigned*)(cbase + r) = pk;
        }
    }
}

// ---------------------------------------------------------------------------
// Pass 3: out[i,:] = d[i] * ( A[i,:] @ supT^T ) + bias
// LDS-free, barrier-free, software-pipelined streaming WMMA kernel.
// ---------------------------------------------------------------------------
struct KStep {
    v4f a0, a1, a2, a3;   // 16 f32 of adjacency (this lane's A-fragment K-runs)
    v4u b[8];             // 4 B fragments x 2 (bf16, K-contiguous)
};

static __device__ __forceinline__ void issue_kstep(
        KStep& f, const float* __restrict__ arow,
        const unsigned short* __restrict__ b0, const unsigned short* __restrict__ b1,
        const unsigned short* __restrict__ b2, const unsigned short* __restrict__ b3,
        int kk, int g) {
    const float* p = arow + kk + 16 * g;
    __builtin_prefetch(p + 128, 0, 0);                  // 4 steps ahead (speculative)
    f.a0 = __builtin_nontemporal_load((const v4f*)(p));       // NT: A streamed once,
    f.a1 = __builtin_nontemporal_load((const v4f*)(p + 4));   // keep L2 for supT
    f.a2 = __builtin_nontemporal_load((const v4f*)(p + 8));
    f.a3 = __builtin_nontemporal_load((const v4f*)(p + 12));
    f.b[0] = *(const v4u*)(b0 + kk);  f.b[1] = *(const v4u*)(b0 + kk + 8);
    f.b[2] = *(const v4u*)(b1 + kk);  f.b[3] = *(const v4u*)(b1 + kk + 8);
    f.b[4] = *(const v4u*)(b2 + kk);  f.b[5] = *(const v4u*)(b2 + kk + 8);
    f.b[6] = *(const v4u*)(b3 + kk);  f.b[7] = *(const v4u*)(b3 + kk + 8);
}

static __device__ __forceinline__ void compute_kstep(
        const KStep& f, v8f& c0, v8f& c1, v8f& c2, v8f& c3) {
    FragAB af, b0, b1, b2, b3;
    af.u[0] = pack_bf16x2(f.a0[0], f.a0[1]);
    af.u[1] = pack_bf16x2(f.a0[2], f.a0[3]);
    af.u[2] = pack_bf16x2(f.a1[0], f.a1[1]);
    af.u[3] = pack_bf16x2(f.a1[2], f.a1[3]);
    af.u[4] = pack_bf16x2(f.a2[0], f.a2[1]);
    af.u[5] = pack_bf16x2(f.a2[2], f.a2[3]);
    af.u[6] = pack_bf16x2(f.a3[0], f.a3[1]);
    af.u[7] = pack_bf16x2(f.a3[2], f.a3[3]);
    b0.q[0] = f.b[0];  b0.q[1] = f.b[1];
    b1.q[0] = f.b[2];  b1.q[1] = f.b[3];
    b2.q[0] = f.b[4];  b2.q[1] = f.b[5];
    b3.q[0] = f.b[6];  b3.q[1] = f.b[7];
    c0 = __builtin_amdgcn_wmma_f32_16x16x32_bf16(false, af.v, false, b0.v,
                                                 (short)0, c0, false, false);
    c1 = __builtin_amdgcn_wmma_f32_16x16x32_bf16(false, af.v, false, b1.v,
                                                 (short)0, c1, false, false);
    c2 = __builtin_amdgcn_wmma_f32_16x16x32_bf16(false, af.v, false, b2.v,
                                                 (short)0, c2, false, false);
    c3 = __builtin_amdgcn_wmma_f32_16x16x32_bf16(false, af.v, false, b3.v,
                                                 (short)0, c3, false, false);
}

__global__ __launch_bounds__(256)
void gcn_gemm_kernel(const float* __restrict__ Adj, const unsigned short* __restrict__ supT,
                     const float* __restrict__ dvec, const float* __restrict__ bias,
                     float* __restrict__ out, int N, int DOUT) {
    const int tid  = threadIdx.x;
    const int lane = tid & 31;
    const int wave = tid >> 5;
    const int wm   = wave & 1;        // 0..1 -> 16-row sub-tile
    const int wn   = wave >> 1;       // 0..3 -> 64-col group
    const int m    = lane & 15;
    const int g    = lane >> 4;

    const int rowA = blockIdx.x * 32 + wm * 16 + m;   // this lane's A row
    const float* arow = Adj + (size_t)rowA * N;

    // B fragment column base pointers (include per-lane 16g K offset)
    const unsigned short* bcol0 = supT + (size_t)(wn * 64 +  0 + m) * N + 16 * g;
    const unsigned short* bcol1 = supT + (size_t)(wn * 64 + 16 + m) * N + 16 * g;
    const unsigned short* bcol2 = supT + (size_t)(wn * 64 + 32 + m) * N + 16 * g;
    const unsigned short* bcol3 = supT + (size_t)(wn * 64 + 48 + m) * N + 16 * g;

    v8f acc0 = {}, acc1 = {}, acc2 = {}, acc3 = {};

    // software pipeline: two in-flight k-steps (N/32 is even; N >= 128)
    KStep fa, fb;
    issue_kstep(fa, arow, bcol0, bcol1, bcol2, bcol3, 0,  g);
    issue_kstep(fb, arow, bcol0, bcol1, bcol2, bcol3, 32, g);
    int kk = 0;
    for (; kk < N - 64; kk += 64) {
        compute_kstep(fa, acc0, acc1, acc2, acc3);
        issue_kstep(fa, arow, bcol0, bcol1, bcol2, bcol3, kk + 64, g);
        compute_kstep(fb, acc0, acc1, acc2, acc3);
        issue_kstep(fb, arow, bcol0, bcol1, bcol2, bcol3, kk + 96, g);
    }
    compute_kstep(fa, acc0, acc1, acc2, acc3);   // kk = N-64
    compute_kstep(fb, acc0, acc1, acc2, acc3);   // kk = N-32

    // epilogue: apply d[row], add bias; C/D tile: M = r + 8g, N = lane&15
    v8f accs[4] = {acc0, acc1, acc2, acc3};
    for (int n = 0; n < 4; n++) {
        int col = wn * 64 + n * 16 + m;
        float bv = bias[col];
        for (int r = 0; r < 8; r++) {
            int row = blockIdx.x * 32 + wm * 16 + g * 8 + r;
            __builtin_nontemporal_store(dvec[row] * accs[n][r] + bv,
                                        out + (size_t)row * DOUT + col);
        }
    }
}

// ---------------------------------------------------------------------------
extern "C" void kernel_launch(void* const* d_in, const int* in_sizes, int n_in,
                              void* d_out, int out_size, void* d_ws, size_t ws_size,
                              hipStream_t stream) {
    const float* adj  = (const float*)d_in[0];
    const float* X    = (const float*)d_in[1];
    const float* W    = (const float*)d_in[2];
    const float* bias = (const float*)d_in[3];
    const int*   l    = (const int*)d_in[4];
    float* out = (float*)d_out;

    const int DOUT = in_sizes[3];                 // 256
    const int DIN  = in_sizes[2] / DOUT;          // 256
    const int N    = in_sizes[1] / DIN;           // 8192

    float* dvec = (float*)d_ws;
    size_t dbytes = ((size_t)N * sizeof(float) + 255) & ~(size_t)255;
    unsigned short* supT = (unsigned short*)((char*)d_ws + dbytes);

    deg_rsqrt_kernel<<<N, 256, 0, stream>>>(adj, l, dvec, N);
    support_gemm_kernel<<<N / 64, 256, 0, stream>>>(X, W, dvec, supT, N, DIN, DOUT);
    gcn_gemm_kernel<<<N / 32, 256, 0, stream>>>(adj, supT, dvec, bias, out, N, DOUT);
}
